// MiMoMoeAttention_68753836474421
// MI455X (gfx1250) — matrix operand order repaired and checked by reading, not compiled, not run on hardware
//
#include <hip/hip_runtime.h>

// ---------------------------------------------------------------------------
// MiMo GQA attention for MI455X (gfx1250, wave32, WMMA).
// Core math path: V_WMMA_F32_16X16X32_BF16 (bf16 A/B, fp32 accumulate).
// GEMMs stage shared B tiles in LDS via the CDNA5 async-to-LDS DMA path
// (ASYNCcnt) when the toolchain exposes it; otherwise a synchronous
// register-bounce staging path is used.
// ---------------------------------------------------------------------------

typedef __bf16 bf16_t;
typedef bf16_t v16bf __attribute__((ext_vector_type(16)));
typedef bf16_t v8bf  __attribute__((ext_vector_type(8)));
typedef float  v8f   __attribute__((ext_vector_type(8)));

#define T_SEQ 2048
#define HID   4096
#define NH_   32
#define NKV_  8
#define DH    128
#define NQKV  6144   // NH*D + 2*NKV*D
#define KDIM  1024   // NKV*D

#if __has_builtin(__builtin_amdgcn_global_load_async_to_lds_b128) && \
    __has_builtin(__builtin_amdgcn_s_wait_asynccnt)
#define USE_ASYNC_LDS 1
#endif

#ifdef USE_ASYNC_LDS
// Builtin prototype (from hipcc diagnostic): v4i addrspace(1)* src,
// v4i addrspace(3)* lds, imm offset, imm cpol.  v4i = gcc-vector int4.
typedef int v4i_vs __attribute__((vector_size(16)));
typedef __attribute__((address_space(1))) v4i_vs gv4i;
typedef __attribute__((address_space(3))) v4i_vs lv4i;
#endif

// ---- WMMA fragment loaders (CDNA5 ISA 7.12.2 layouts, wave32) --------------
// A (16x32 bf16, row-major src, ld in elements):
//   lanes 0-15: M=lane, K = {0..7, 16..23}; lanes 16-31: M=lane-16, K={8..15, 24..31}
__device__ __forceinline__ v16bf load_a_frag(const bf16_t* p, int ld) {
  const int lane = threadIdx.x & 31;
  const int m    = lane & 15;
  const int koff = (lane >> 4) << 3;           // 0 or 8
  const bf16_t* q = p + (size_t)m * ld + koff;
  v8bf lo = *(const v8bf*)(q);                 // K base+0..7
  v8bf hi = *(const v8bf*)(q + 16);            // K base+16..23
  return __builtin_shufflevector(lo, hi, 0,1,2,3,4,5,6,7,8,9,10,11,12,13,14,15);
}

// B (32x16 bf16) from N-major (transposed) storage, ld in elements:
//   lanes 0-15: N=lane, K = {0..7, 16..23}; lanes 16-31: N=lane-16, K={8..15, 24..31}
__device__ __forceinline__ v16bf load_b_frag(const bf16_t* p, int ld) {
  const int lane = threadIdx.x & 31;
  const int n    = lane & 15;
  const int koff = (lane >> 4) << 3;           // 0 or 8
  const bf16_t* q = p + (size_t)n * ld + koff;
  v8bf lo = *(const v8bf*)(q);
  v8bf hi = *(const v8bf*)(q + 16);
  return __builtin_shufflevector(lo, hi, 0,1,2,3,4,5,6,7,8,9,10,11,12,13,14,15);
}

__device__ __forceinline__ v8f wmma_bf16(v16bf a, v16bf b, v8f c) {
  return __builtin_amdgcn_wmma_f32_16x16x32_bf16(false, a, false, b, (short)0, c,
                                                 false, false);
}

// Stage 16 bytes global -> LDS (per thread). Async (ASYNCcnt path) if available.
__device__ __forceinline__ void stage16(const bf16_t* gsrc, bf16_t* ldst) {
#ifdef USE_ASYNC_LDS
  __builtin_amdgcn_global_load_async_to_lds_b128((gv4i*)(void*)gsrc,
                                                 (lv4i*)(void*)ldst, 0, 0);
#else
  *(v8bf*)ldst = *(const v8bf*)gsrc;
#endif
}

// ---- small prep kernels ----------------------------------------------------
__global__ __launch_bounds__(256) void cvt_bf16_kernel(const float* __restrict__ src,
                                                       bf16_t* __restrict__ dst, int n) {
  int i = blockIdx.x * 256 + threadIdx.x;
  if (i < n) dst[i] = (bf16_t)src[i];
}

// src: rows x cols fp32 row-major -> dst: cols x rows bf16 row-major
__global__ __launch_bounds__(256) void transpose_f32_bf16(const float* __restrict__ src,
                                                          bf16_t* __restrict__ dst,
                                                          int rows, int cols) {
  __shared__ float tile[32][33];
  const int tx = threadIdx.x;                  // 0..31
  const int ty = threadIdx.y;                  // 0..7
  const int c0 = blockIdx.x * 32;
  const int r0 = blockIdx.y * 32;
#pragma unroll
  for (int i = 0; i < 32; i += 8)
    tile[ty + i][tx] = src[(size_t)(r0 + ty + i) * cols + c0 + tx];
  __syncthreads();
#pragma unroll
  for (int i = 0; i < 32; i += 8)
    dst[(size_t)(c0 + ty + i) * rows + r0 + tx] = (bf16_t)tile[tx][ty + i];
}

__global__ __launch_bounds__(256) void bias_cat_kernel(const float* __restrict__ bq,
                                                       const float* __restrict__ bk,
                                                       const float* __restrict__ bv,
                                                       float* __restrict__ o) {
  int i = blockIdx.x * 256 + threadIdx.x;
  if (i < 4096)       o[i] = bq[i];
  else if (i < 5120)  o[i] = bk[i - 4096];
  else if (i < 6144)  o[i] = bv[i - 5120];
}

// ---- bf16 WMMA GEMM: C(MxN,f32) = A(MxK,bf16) * Bt(NxK,bf16)^T + bias ------
// grid: (N/64, M/256), block: 256 (8 waves). Block tile 256x64; wave tile 32x64.
// The 64x32 B tile is staged once per block into LDS (async DMA, double
// buffered) and shared by all 8 waves; A fragments stream from global/L2.
__global__ __launch_bounds__(256) void gemm_bf16_kernel(const bf16_t* __restrict__ A,
                                                        const bf16_t* __restrict__ Bt,
                                                        const float* __restrict__ bias,
                                                        float* __restrict__ C,
                                                        int M, int N, int K) {
  __shared__ bf16_t lds_b[2][64 * 32];         // 2 x 4 KiB double buffer
  const int tid  = threadIdx.x;
  const int wave = tid >> 5;
  const int lane = tid & 31;
  const int m0 = blockIdx.y * 256 + wave * 32;
  const int n0 = blockIdx.x * 64;
  const bf16_t* a = A + (size_t)m0 * K;
  // staging assignment: thread -> (row n, 16B chunk of k)
  const int sn = tid >> 2;                     // 0..63
  const int sk = (tid & 3) << 3;               // 0,8,16,24
  const bf16_t* gstage = Bt + (size_t)(n0 + sn) * K + sk;

  v8f acc[8] = {};                             // [0..3]: rows m0.., [4..7]: rows m0+16..
  const int nsteps = K >> 5;

  stage16(gstage, &lds_b[0][sn * 32 + sk]);    // prologue: buffer 0 <- k=0
  for (int s = 0; s < nsteps; ++s) {
    const int k = s << 5;
    const int buf = s & 1;
    if (s + 1 < nsteps) {                      // kick next tile's DMA early
      stage16(gstage + (k + 32), &lds_b[buf ^ 1][sn * 32 + sk]);
#ifdef USE_ASYNC_LDS
      __builtin_amdgcn_s_wait_asynccnt(1);     // current buffer complete
#endif
    } else {
#ifdef USE_ASYNC_LDS
      __builtin_amdgcn_s_wait_asynccnt(0);
#endif
    }
    __syncthreads();                           // all waves' staging visible

    v16bf a0 = load_a_frag(a + k, K);
    v16bf a1 = load_a_frag(a + (size_t)16 * K + k, K);
#pragma unroll
    for (int j = 0; j < 4; ++j) {
      v16bf bf = load_b_frag(&lds_b[buf][j * 16 * 32], 32);
      acc[j]     = wmma_bf16(a0, bf, acc[j]);
      acc[j + 4] = wmma_bf16(a1, bf, acc[j + 4]);
    }
    __syncthreads();                           // done reading before next overwrite
  }

  const int half = lane >> 4, ln = lane & 15;
#pragma unroll
  for (int j = 0; j < 4; ++j)
#pragma unroll
    for (int r = 0; r < 8; ++r) {
      const int nn = n0 + j * 16 + ln;
      const float bb = bias ? bias[nn] : 0.f;
      const int mm = m0 + r + 8 * half;        // C layout: VGPR r -> row r (+8 hi half)
      C[(size_t)mm * N + nn]        = acc[j][r] + bb;
      C[(size_t)(mm + 16) * N + nn] = acc[j + 4][r] + bb;
    }
}

// ---- RoPE + scatter kernels ------------------------------------------------
// ln(1e6)/64
#define ROPE_LN_OVER_HALF 0.21586735246819178f

__global__ __launch_bounds__(256) void rope_q_kernel(const int* __restrict__ pos,
                                                     const float* __restrict__ qkv,
                                                     bf16_t* __restrict__ qb) {
  int idx = blockIdx.x * 256 + threadIdx.x;    // T*NH*64 threads
  const int i = idx & 63;
  const int h = (idx >> 6) & 31;
  const int t = idx >> 11;
  const float* src = qkv + (size_t)t * NQKV + h * DH;
  const float x1 = src[i], x2 = src[i + 64];
  const float f = (float)pos[t] * __expf(-(float)i * ROPE_LN_OVER_HALF);
  const float c = cosf(f), s = sinf(f);
  bf16_t* dst = qb + (size_t)t * HID + h * DH;
  dst[i]      = (bf16_t)(x1 * c - x2 * s);
  dst[i + 64] = (bf16_t)(x2 * c + x1 * s);
}

__global__ __launch_bounds__(256) void rope_k_kernel(const int* __restrict__ pos,
                                                     const float* __restrict__ qkv,
                                                     bf16_t* __restrict__ kb,
                                                     float* __restrict__ kv0) {
  int idx = blockIdx.x * 256 + threadIdx.x;    // T*NKV*64 threads
  const int i = idx & 63;
  const int h = (idx >> 6) & 7;
  const int t = idx >> 9;
  const float* src = qkv + (size_t)t * NQKV + HID + h * DH;
  const float x1 = src[i], x2 = src[i + 64];
  const float f = (float)pos[t] * __expf(-(float)i * ROPE_LN_OVER_HALF);
  const float c = cosf(f), s = sinf(f);
  const float o1 = x1 * c - x2 * s;
  const float o2 = x2 * c + x1 * s;
  bf16_t* kd = kb + (size_t)t * KDIM + h * DH;
  kd[i]      = (bf16_t)o1;
  kd[i + 64] = (bf16_t)o2;
  float* kf = kv0 + ((size_t)t * NKV_ + h) * DH;  // kv_fused[0]
  kf[i]      = o1;
  kf[i + 64] = o2;
}

__global__ __launch_bounds__(256) void v_scatter_kernel(const float* __restrict__ qkv,
                                                        bf16_t* __restrict__ vt,
                                                        float* __restrict__ kv1) {
  int idx = blockIdx.x * 256 + threadIdx.x;    // T*NKV*128 threads
  const int d = idx & 127;
  const int h = (idx >> 7) & 7;
  const int t = idx >> 10;
  const float v = qkv[(size_t)t * NQKV + HID + KDIM + h * DH + d];
  kv1[((size_t)t * NKV_ + h) * DH + d] = v;                    // kv_fused[1]
  vt[((size_t)h * DH + d) * T_SEQ + t] = (bf16_t)v;            // V^T, d-major
}

// ---- flash attention: one wave per (16-query block, head) ------------------
__global__ __launch_bounds__(32) void attn_kernel(const bf16_t* __restrict__ qb,
                                                  const bf16_t* __restrict__ kb,
                                                  const bf16_t* __restrict__ vt,
                                                  bf16_t* __restrict__ ao) {
  __shared__ bf16_t lds_p[16 * 32];            // P tile staging (C->A re-fragment)
  const int lane = threadIdx.x;
  const int half = lane >> 4;
  const int ln   = lane & 15;
  const int qblk = blockIdx.x;                 // 0..127
  const int h    = blockIdx.y;                 // 0..31
  const int hk   = h >> 2;                     // GQA: G=4

  const bf16_t* qbase = qb + (size_t)(qblk * 16) * HID + h * DH;
  v16bf aq[4];
#pragma unroll
  for (int kk = 0; kk < 4; ++kk) aq[kk] = load_a_frag(qbase + kk * 32, HID);

  v8f acc[8] = {};
  float mrow[8], lrow[8];
#pragma unroll
  for (int r = 0; r < 8; ++r) { mrow[r] = -1e30f; lrow[r] = 0.f; }

  const float scale = 0.08838834764831845f;    // 1/sqrt(128)
  const int row_hi = qblk * 16 + 15;
  const int nch = (row_hi >> 5) + 1;           // causal: key chunks of 32
  for (int c = 0; c < nch; ++c) {
    const int s0 = c << 5;
    const bf16_t* kb0 = kb + (size_t)s0 * KDIM + hk * DH;
    v8f sc0 = {}, sc1 = {};
#pragma unroll
    for (int kk = 0; kk < 4; ++kk)
      sc0 = wmma_bf16(aq[kk], load_b_frag(kb0 + kk * 32, KDIM), sc0);
#pragma unroll
    for (int kk = 0; kk < 4; ++kk)
      sc1 = wmma_bf16(aq[kk], load_b_frag(kb0 + (size_t)16 * KDIM + kk * 32, KDIM), sc1);

    float p0[8], p1[8];
#pragma unroll
    for (int r = 0; r < 8; ++r) {
      const int rowg = qblk * 16 + r + 8 * half;
      float x0 = sc0[r] * scale;
      float x1 = sc1[r] * scale;
      if (s0 + ln > rowg)      x0 = -1e30f;    // causal mask
      if (s0 + 16 + ln > rowg) x1 = -1e30f;
      float mx = fmaxf(x0, x1);                // row-max over this half's 16 lanes
      mx = fmaxf(mx, __shfl_xor(mx, 1));
      mx = fmaxf(mx, __shfl_xor(mx, 2));
      mx = fmaxf(mx, __shfl_xor(mx, 4));
      mx = fmaxf(mx, __shfl_xor(mx, 8));
      const float mnew  = fmaxf(mrow[r], mx);
      const float alpha = __expf(mrow[r] - mnew);
      mrow[r] = mnew;
      const float e0 = __expf(x0 - mnew);
      const float e1 = __expf(x1 - mnew);
      float sm = e0 + e1;
      sm += __shfl_xor(sm, 1);
      sm += __shfl_xor(sm, 2);
      sm += __shfl_xor(sm, 4);
      sm += __shfl_xor(sm, 8);
      lrow[r] = lrow[r] * alpha + sm;
#pragma unroll
      for (int nc = 0; nc < 8; ++nc) acc[nc][r] *= alpha;
      p0[r] = e0; p1[r] = e1;
    }

    __syncthreads();                           // protect LDS reuse across iterations
#pragma unroll
    for (int r = 0; r < 8; ++r) {
      const int rr = r + 8 * half;
      lds_p[rr * 32 + ln]      = (bf16_t)p0[r];
      lds_p[rr * 32 + 16 + ln] = (bf16_t)p1[r];
    }
    __syncthreads();
    v16bf ap = load_a_frag(lds_p, 32);         // P as 16x32 A fragment
    const bf16_t* vb = vt + (size_t)hk * DH * T_SEQ + s0;
#pragma unroll
    for (int nc = 0; nc < 8; ++nc)
      acc[nc] = wmma_bf16(ap, load_b_frag(vb + (size_t)(nc * 16) * T_SEQ, T_SEQ),
                          acc[nc]);
  }

#pragma unroll
  for (int r = 0; r < 8; ++r) lrow[r] = 1.0f / lrow[r];
#pragma unroll
  for (int nc = 0; nc < 8; ++nc)
#pragma unroll
    for (int r = 0; r < 8; ++r) {
      const int tt = qblk * 16 + r + 8 * half;
      ao[(size_t)tt * HID + h * DH + nc * 16 + ln] = (bf16_t)(acc[nc][r] * lrow[r]);
    }
}

// ---------------------------------------------------------------------------
extern "C" void kernel_launch(void* const* d_in, const int* in_sizes, int n_in,
                              void* d_out, int out_size, void* d_ws, size_t ws_size,
                              hipStream_t stream) {
  (void)in_sizes; (void)n_in; (void)out_size; (void)ws_size;
  const int*   pos    = (const int*)d_in[0];
  const float* hidden = (const float*)d_in[1];
  const float* Wq = (const float*)d_in[2];
  const float* bq = (const float*)d_in[3];
  const float* Wk = (const float*)d_in[4];
  const float* bk = (const float*)d_in[5];
  const float* Wv = (const float*)d_in[6];
  const float* bv = (const float*)d_in[7];
  const float* Wo = (const float*)d_in[8];
  float* out = (float*)d_out;

  // workspace layout (bytes)
  char* ws = (char*)d_ws;
  bf16_t* hid_bf = (bf16_t*)(ws + 0);                    // 16 MiB (reused as attn bf16)
  bf16_t* wcat   = (bf16_t*)(ws + 16777216);             // 48 MiB: [WqT;WkT;WvT] N=6144 x K=4096
  bf16_t* wot    = (bf16_t*)(ws + 67108864);             // 32 MiB: WoT 4096x4096
  float*  qkv    = (float*)(ws + 100663296);             // 48 MiB: (2048 x 6144) f32
  bf16_t* q_bf   = (bf16_t*)(ws + 150994944);            // 16 MiB
  bf16_t* k_bf   = (bf16_t*)(ws + 167772160);            // 4 MiB
  bf16_t* v_t    = (bf16_t*)(ws + 171966464);            // 4 MiB  (NKV, D, T)
  float*  biasc  = (float*)(ws + 176160768);             // 24 KiB
  bf16_t* attn_bf = hid_bf;                              // alias (hidden dead after QKV GEMM)

  float* kv0 = out + (size_t)T_SEQ * HID;                          // kv_fused[0] (k)
  float* kv1 = kv0 + (size_t)T_SEQ * NKV_ * DH;                    // kv_fused[1] (v)

  const dim3 tb(32, 8);

  // 1) precision conversion + weight transposes (N-major bf16 for B fragments)
  cvt_bf16_kernel<<<32768, 256, 0, stream>>>(hidden, hid_bf, T_SEQ * HID);
  transpose_f32_bf16<<<dim3(128, 128), tb, 0, stream>>>(Wq, wcat, 4096, 4096);
  transpose_f32_bf16<<<dim3(32, 128), tb, 0, stream>>>(Wk, wcat + (size_t)4096 * 4096, 4096, 1024);
  transpose_f32_bf16<<<dim3(32, 128), tb, 0, stream>>>(Wv, wcat + (size_t)5120 * 4096, 4096, 1024);
  transpose_f32_bf16<<<dim3(128, 128), tb, 0, stream>>>(Wo, wot, 4096, 4096);
  bias_cat_kernel<<<24, 256, 0, stream>>>(bq, bk, bv, biasc);

  // 2) fused QKV projection: (2048 x 4096) @ (4096 x 6144) -> qkv f32
  gemm_bf16_kernel<<<dim3(96, 8), 256, 0, stream>>>(hid_bf, wcat, biasc, qkv,
                                                    T_SEQ, NQKV, HID);

  // 3) RoPE + kv_fused outputs + attention operand staging
  rope_q_kernel<<<16384, 256, 0, stream>>>(pos, qkv, q_bf);        // T*NH*64
  rope_k_kernel<<<4096, 256, 0, stream>>>(pos, qkv, k_bf, kv0);    // T*NKV*64
  v_scatter_kernel<<<8192, 256, 0, stream>>>(qkv, v_t, kv1);       // T*NKV*128

  // 4) causal flash attention (WMMA scores + WMMA PV, online softmax)
  attn_kernel<<<dim3(T_SEQ / 16, NH_), 32, 0, stream>>>(q_bf, k_bf, v_t, attn_bf);

  // 5) output projection: (2048 x 4096) @ (4096 x 4096) -> out f32
  gemm_bf16_kernel<<<dim3(64, 8), 256, 0, stream>>>(attn_bf, wot, nullptr, out,
                                                    T_SEQ, HID, HID);
}
